// RelMarkerv0_83348135346545
// MI455X (gfx1250) — matrix operand order repaired and checked by reading, not compiled, not run on hardware
//
#include <hip/hip_runtime.h>
#include <stdint.h>
#include <stddef.h>

// ---------------------------------------------------------------------------
// Types for CDNA5 WMMA (wave32, 16x16x32 bf16 -> f32 accumulate)
// ---------------------------------------------------------------------------
typedef __bf16 bf16_t;
typedef bf16_t v8bf  __attribute__((ext_vector_type(8)));
typedef bf16_t v16bf __attribute__((ext_vector_type(16)));
typedef float  v8f   __attribute__((ext_vector_type(8)));

// Problem dims (from reference): B=8, L=512, D=768, H=4D=3072, N=64 spans
static constexpr int Bb    = 8;
static constexpr int Ll    = 512;
static constexpr int Dd    = 768;
static constexpr int Hh    = 3072;   // 4*D
static constexpr int NS    = 64;     // spans per batch
static constexpr int NPAIR = NS * (NS - 1);        // 4032 ordered pairs/batch
static constexpr int ROWS  = Bb * NPAIR;           // 32256 pair rows total
static constexpr int CHUNK = 2016;                 // pair rows per chunk
static constexpr int NCHUNK = ROWS / CHUNK;        // 16

// ---------------------------------------------------------------------------
// Transpose-and-convert: W (K x N, f32, row-major) -> Wt (N x K, bf16)
// K,N are multiples of 32 for every weight here.
// ---------------------------------------------------------------------------
__global__ __launch_bounds__(256)
void transpose_to_bf16(const float* __restrict__ W, bf16_t* __restrict__ Wt,
                       int K, int N)
{
    __shared__ float tile[32][33];
    const int n0 = blockIdx.x * 32;
    const int k0 = blockIdx.y * 32;
    const int tx = threadIdx.x & 31;
    const int ty = threadIdx.x >> 5;      // 0..7
    #pragma unroll
    for (int i = 0; i < 32; i += 8)
        tile[ty + i][tx] = W[(size_t)(k0 + ty + i) * N + (n0 + tx)];
    __syncthreads();
    #pragma unroll
    for (int i = 0; i < 32; i += 8)
        Wt[(size_t)(n0 + ty + i) * K + (k0 + tx)] = (bf16_t)tile[tx][ty + i];
}

// ---------------------------------------------------------------------------
// Gather h rows at span start/end indices, convert to bf16.
// ---------------------------------------------------------------------------
__global__ __launch_bounds__(256)
void gather_rows_bf16(const float* __restrict__ h, const int* __restrict__ span,
                      bf16_t* __restrict__ outS, bf16_t* __restrict__ outE)
{
    const int r  = blockIdx.x;            // b*64 + n
    const int b  = r >> 6;
    const int i0 = span[r * 2 + 0];
    const int i1 = span[r * 2 + 1];
    const float* ps = h + (size_t)(b * Ll + i0) * Dd;
    const float* pe = h + (size_t)(b * Ll + i1) * Dd;
    for (int d = threadIdx.x; d < Dd; d += blockDim.x) {
        outS[(size_t)r * Dd + d] = (bf16_t)ps[d];
        outE[(size_t)r * Dd + d] = (bf16_t)pe[d];
    }
}

// ---------------------------------------------------------------------------
// bf16 WMMA GEMM, double-buffered LDS software pipeline.
//   MODE 0: A is M x K bf16 row-major (lda)
//   MODE 1: A is the 512 x 768 entity table; logical row rowBase+m is
//           concat(ent[b,i], ent[b,j]) decoded from the pair index (hoisted
//           out of the K loop: two base pointers per staging chunk).
//   Bt: N x K bf16 (B transposed) -> contiguous fragment rows.
// Block = 256 threads = 8 wave32s; tile 128x128; K staged 32/iter.
// Wave grid 4(M) x 2(N): 2x4 fragments of 16x16 -> 8 WMMAs / wave / stage.
// LDS rows padded to 40 bf16 (80B): conflict-free, 16B-aligned ds_load_b128.
// Pipeline: global_load(next) -> WMMA(cur) -> ds_store(next) -> barrier.
// ---------------------------------------------------------------------------
template<int MODE>
__global__ __launch_bounds__(256)
void gemm_bf16_wmma(const bf16_t* __restrict__ A, int lda,
                    const bf16_t* __restrict__ Bt,
                    const float* __restrict__ bias,
                    float* __restrict__ Cf, bf16_t* __restrict__ Cb, int ldc,
                    int M, int N, int K, int relu, int rowBase)
{
    __shared__ bf16_t As[2][128 * 40];
    __shared__ bf16_t Bs[2][128 * 40];

    const int tid   = threadIdx.x;
    const int lane  = tid & 31;
    const int wid   = tid >> 5;
    const int mbase = (wid >> 1) * 32;    // wave M offset within tile
    const int nbase = (wid & 1) * 64;     // wave N offset within tile
    const int tileN = blockIdx.x * 128;
    const int tileM = blockIdx.y * 128;

    const int lh = lane & 15;
    const int hb = (lane >> 4) << 3;      // A fragment K-half offset (0/8)
    const int kb = (lane >> 4) << 3;      // B fragment K-half offset (0/8)

    // ---- per-thread staging descriptors: 2 chunks of 8 bf16 (A and B) ----
    const bf16_t* aF[2];                  // base ptr, first half (or direct)
    const bf16_t* aS[2];                  // base ptr, second half (MODE 1)
    const bf16_t* bP[2];
    int  ldsOff[2], kcA[2];
    bool aValid[2];
    #pragma unroll
    for (int i = 0; i < 2; ++i) {
        const int cid = tid + i * 256;
        const int r   = cid >> 2;
        const int kc  = (cid & 3) << 3;
        ldsOff[i] = r * 40 + kc;
        kcA[i]    = kc;
        const int grow = tileM + r;
        aValid[i] = grow < M;
        const int crow = aValid[i] ? grow : 0;   // clamp for safe decode
        if (MODE == 0) {
            aF[i] = A + (size_t)crow * lda + kc;
            aS[i] = aF[i];
        } else {
            const int g  = rowBase + crow;
            const int b  = g / NPAIR;
            const int p  = g - b * NPAIR;
            const int ii = p / (NS - 1);
            const int t  = p - ii * (NS - 1);
            const int jj = t + (t >= ii ? 1 : 0);
            aF[i] = A + (size_t)(b * NS + ii) * Dd + kc;        // k in [0,768)
            aS[i] = A + (size_t)(b * NS + jj) * Dd + kc - Dd;   // k in [768,1536)
        }
        bP[i] = Bt + (size_t)(tileN + r) * K + kc;
    }

    auto loadStage = [&](int k0, v8bf va[2], v8bf vb[2]) {
        #pragma unroll
        for (int i = 0; i < 2; ++i) {
            v8bf v = {};
            if (aValid[i]) {
                const bf16_t* p = (MODE == 0 || (k0 + kcA[i]) < Dd)
                                  ? (aF[i] + k0) : (aS[i] + k0);
                v = *(const v8bf*)p;
            }
            va[i] = v;
            vb[i] = *(const v8bf*)(bP[i] + k0);
        }
    };
    auto storeStage = [&](int buf, const v8bf va[2], const v8bf vb[2]) {
        #pragma unroll
        for (int i = 0; i < 2; ++i) {
            *(v8bf*)&As[buf][ldsOff[i]] = va[i];
            *(v8bf*)&Bs[buf][ldsOff[i]] = vb[i];
        }
    };

    v8f acc[2][4] = {};                   // 2 M-frags x 4 N-frags

    auto compute = [&](int buf) {
        v16bf afrag[2];
        #pragma unroll
        for (int mi = 0; mi < 2; ++mi) {
            const int r = mbase + mi * 16 + lh;
            v8bf lo = *(const v8bf*)&As[buf][r * 40 + hb];
            v8bf hi = *(const v8bf*)&As[buf][r * 40 + 16 + hb];
            afrag[mi] = __builtin_shufflevector(lo, hi,
                0,1,2,3,4,5,6,7,8,9,10,11,12,13,14,15);
        }
        #pragma unroll
        for (int ni = 0; ni < 4; ++ni) {
            const int rn = nbase + ni * 16 + lh;
            v8bf lo = *(const v8bf*)&Bs[buf][rn * 40 + kb];
            v8bf hi = *(const v8bf*)&Bs[buf][rn * 40 + 16 + kb];
            v16bf bfrag = __builtin_shufflevector(lo, hi,
                0,1,2,3,4,5,6,7,8,9,10,11,12,13,14,15);
            acc[0][ni] = __builtin_amdgcn_wmma_f32_16x16x32_bf16(
                false, afrag[0], false, bfrag, (short)0, acc[0][ni], false, false);
            acc[1][ni] = __builtin_amdgcn_wmma_f32_16x16x32_bf16(
                false, afrag[1], false, bfrag, (short)0, acc[1][ni], false, false);
        }
    };

    // ---- software pipeline: prime stage 0, then overlap ----
    v8bf va[2], vb[2];
    loadStage(0, va, vb);
    storeStage(0, va, vb);
    __syncthreads();

    int buf = 0;
    for (int k0 = 0; k0 < K; k0 += 32) {
        const bool hasNext = (k0 + 32) < K;
        v8bf na[2], nb[2];
        if (hasNext) loadStage(k0 + 32, na, nb);   // issue global loads
        compute(buf);                              // WMMAs hide load latency
        if (hasNext) storeStage(buf ^ 1, na, nb);  // ds_store after compute
        __syncthreads();
        buf ^= 1;
    }

    // ---- epilogue: bias + optional ReLU, store f32 or bf16 ----
    #pragma unroll
    for (int mi = 0; mi < 2; ++mi) {
        const int rb = tileM + mbase + mi * 16 + ((lane >> 4) << 3);
        #pragma unroll
        for (int ni = 0; ni < 4; ++ni) {
            const int col = tileN + nbase + ni * 16 + lh;
            const float bv = bias ? bias[col] : 0.0f;
            #pragma unroll
            for (int v = 0; v < 8; ++v) {
                const int row = rb + v;
                if (row >= M) continue;
                float x = acc[mi][ni][v] + bv;
                if (relu) x = fmaxf(x, 0.0f);
                if (Cf) Cf[(size_t)row * ldc + col] = x;
                else    Cb[(size_t)row * ldc + col] = (bf16_t)x;
            }
        }
    }
}

// ---------------------------------------------------------------------------
// Host side
// ---------------------------------------------------------------------------
extern "C" void kernel_launch(void* const* d_in, const int* in_sizes, int n_in,
                              void* d_out, int out_size, void* d_ws, size_t ws_size,
                              hipStream_t stream)
{
    (void)in_sizes; (void)n_in; (void)out_size; (void)ws_size;
    const float* h    = (const float*)d_in[0];
    const int*   span = (const int*)d_in[1];
    const float* Ws1  = (const float*)d_in[2];
    const float* bs1  = (const float*)d_in[3];
    const float* Ws2  = (const float*)d_in[4];
    const float* bs2  = (const float*)d_in[5];
    const float* We1  = (const float*)d_in[6];
    const float* be1  = (const float*)d_in[7];
    const float* We2  = (const float*)d_in[8];
    const float* be2  = (const float*)d_in[9];
    const float* Wo1  = (const float*)d_in[10];
    const float* bo1  = (const float*)d_in[11];
    const float* Wo2  = (const float*)d_in[12];
    const float* bo2  = (const float*)d_in[13];
    const float* Wr1  = (const float*)d_in[14];
    const float* br1  = (const float*)d_in[15];
    const float* Wr2  = (const float*)d_in[16];
    const float* br2  = (const float*)d_in[17];

    uint8_t* ws = (uint8_t*)d_ws;
    size_t off = 0;
    auto alloc = [&](size_t bytes) -> void* {
        void* p = ws + off;
        off = (off + bytes + 255) & ~(size_t)255;
        return p;
    };
    // bf16 transposed weights (~72 MB, L2-resident thereafter)
    bf16_t* Wt_s1 = (bf16_t*)alloc((size_t)Hh * Dd * 2);
    bf16_t* Wt_s2 = (bf16_t*)alloc((size_t)Dd * Hh * 2);
    bf16_t* Wt_e1 = (bf16_t*)alloc((size_t)Hh * Dd * 2);
    bf16_t* Wt_e2 = (bf16_t*)alloc((size_t)Dd * Hh * 2);
    bf16_t* Wt_o1 = (bf16_t*)alloc((size_t)(8*Dd) * (2*Dd) * 2);
    bf16_t* Wt_o2 = (bf16_t*)alloc((size_t)Dd * (8*Dd) * 2);
    bf16_t* Wt_r1 = (bf16_t*)alloc((size_t)(8*Dd) * (2*Dd) * 2);
    bf16_t* Wt_r2 = (bf16_t*)alloc((size_t)Dd * (8*Dd) * 2);
    // activations
    bf16_t* hs     = (bf16_t*)alloc((size_t)512 * Dd * 2);
    bf16_t* he     = (bf16_t*)alloc((size_t)512 * Dd * 2);
    bf16_t* hidS   = (bf16_t*)alloc((size_t)512 * Hh * 2);
    bf16_t* hidE   = (bf16_t*)alloc((size_t)512 * Hh * 2);
    bf16_t* catsp  = (bf16_t*)alloc((size_t)512 * (2*Dd) * 2);
    bf16_t* enthid = (bf16_t*)alloc((size_t)512 * (8*Dd) * 2);
    bf16_t* ent    = (bf16_t*)alloc((size_t)512 * Dd * 2);
    bf16_t* phid   = (bf16_t*)alloc((size_t)CHUNK * (8*Dd) * 2);

    // --- convert weights to transposed bf16 (once per launch) ---
    transpose_to_bf16<<<dim3(Hh/32,   Dd/32),   256, 0, stream>>>(Ws1, Wt_s1, Dd,   Hh);
    transpose_to_bf16<<<dim3(Dd/32,   Hh/32),   256, 0, stream>>>(Ws2, Wt_s2, Hh,   Dd);
    transpose_to_bf16<<<dim3(Hh/32,   Dd/32),   256, 0, stream>>>(We1, Wt_e1, Dd,   Hh);
    transpose_to_bf16<<<dim3(Dd/32,   Hh/32),   256, 0, stream>>>(We2, Wt_e2, Hh,   Dd);
    transpose_to_bf16<<<dim3(8*Dd/32, 2*Dd/32), 256, 0, stream>>>(Wo1, Wt_o1, 2*Dd, 8*Dd);
    transpose_to_bf16<<<dim3(Dd/32,   8*Dd/32), 256, 0, stream>>>(Wo2, Wt_o2, 8*Dd, Dd);
    transpose_to_bf16<<<dim3(8*Dd/32, 2*Dd/32), 256, 0, stream>>>(Wr1, Wt_r1, 2*Dd, 8*Dd);
    transpose_to_bf16<<<dim3(Dd/32,   8*Dd/32), 256, 0, stream>>>(Wr2, Wt_r2, 8*Dd, Dd);

    // --- gather span rows FIRST (512 rows, not 4096) ---
    gather_rows_bf16<<<512, 256, 0, stream>>>(h, span, hs, he);

    auto gemm0 = [&](const bf16_t* A, int lda, const bf16_t* Bt, const float* bias,
                     float* Cf, bf16_t* Cb, int ldc, int M, int N, int K, int relu) {
        dim3 g(N / 128, (M + 127) / 128);
        gemm_bf16_wmma<0><<<g, 256, 0, stream>>>(A, lda, Bt, bias, Cf, Cb, ldc,
                                                 M, N, K, relu, 0);
    };
    auto gemm1 = [&](const bf16_t* ent_tab, const bf16_t* Bt, const float* bias,
                     bf16_t* Cb, int ldc, int M, int N, int K, int rowBase) {
        dim3 g(N / 128, (M + 127) / 128);
        gemm_bf16_wmma<1><<<g, 256, 0, stream>>>(ent_tab, 0, Bt, bias,
                                                 nullptr, Cb, ldc,
                                                 M, N, K, 1, rowBase);
    };

    // --- start/end MLPs on gathered rows; ReLU(cat) fused into layer-2 ---
    gemm0(hs,   Dd, Wt_s1, bs1, nullptr, hidS,       Hh,   512, Hh, Dd, 1);
    gemm0(hidS, Hh, Wt_s2, bs2, nullptr, catsp,      2*Dd, 512, Dd, Hh, 1);
    gemm0(he,   Dd, Wt_e1, be1, nullptr, hidE,       Hh,   512, Hh, Dd, 1);
    gemm0(hidE, Hh, Wt_e2, be2, nullptr, catsp + Dd, 2*Dd, 512, Dd, Hh, 1);

    // --- entity MLP ---
    gemm0(catsp,  2*Dd, Wt_o1, bo1, nullptr, enthid, 8*Dd, 512, 8*Dd, 2*Dd, 1);
    gemm0(enthid, 8*Dd, Wt_o2, bo2, nullptr, ent,    Dd,   512, Dd,   8*Dd, 0);

    // --- pair MLP, chunked; A gathered on the fly from the entity table ---
    for (int c = 0; c < NCHUNK; ++c) {
        gemm1(ent, Wt_r1, br1, phid, 8*Dd, CHUNK, 8*Dd, 2*Dd, c * CHUNK);
        gemm0(phid, 8*Dd, Wt_r2, br2,
              (float*)d_out + (size_t)c * CHUNK * Dd, nullptr, Dd,
              CHUNK, Dd, 8*Dd, 0);
    }
}
